// GraphTransformerModule_11063835754632
// MI455X (gfx1250) — matrix operand order repaired and checked by reading, not compiled, not run on hardware
//
#include <hip/hip_runtime.h>

typedef __attribute__((ext_vector_type(16))) _Float16 v16h;
typedef __attribute__((ext_vector_type(8)))  float    v8f;

#define DEVI __device__ __forceinline__

// LDS tile geometry: 128 rows x 128 halves, padded to 136 halves (272B = 17 uint4)
// per row: keeps 16B alignment for b128 fragment loads and rotates LDS banks.
#define LDT   136
#define LDTU4 17

#if defined(__has_builtin)
# if __has_builtin(__builtin_amdgcn_global_load_async_to_lds_b128) && \
     __has_builtin(__builtin_amdgcn_s_wait_asynccnt)
#  define HAVE_ASYNC_LDS 1
# endif
#endif
#ifndef HAVE_ASYNC_LDS
# define HAVE_ASYNC_LDS 0
#endif

#if HAVE_ASYNC_LDS
// builtin prototype is (v4i addrspace(1)*, v4i addrspace(3)*, imm int, imm int)
typedef int i32x4_av __attribute__((vector_size(16)));
typedef __attribute__((address_space(1))) i32x4_av* as1_i32x4;
typedef __attribute__((address_space(3))) i32x4_av* as3_i32x4;
#endif

// ---------------- WMMA fragment helpers (CDNA5 16x16x32 f16, wave32) ----------------
// A (16x32 f16): lanes 0-15 -> row M=lane, K in {0..7,16..23}; lanes 16-31 -> row
// M=lane-16, K in {8..15,24..31}. Each lane's halves form two contiguous 8-half runs
// => two b128 LDS loads.
DEVI v16h load_a_frag(const _Float16* A, int lda, int lane) {
  int m  = lane & 15;
  int kq = (lane >> 4) << 3;                 // 0 or 8
  const _Float16* base = A + (size_t)m * lda + kq;
  union { v16h h; uint4 q[2]; } f;
  f.q[0] = *(const uint4*)(base);            // K = kq .. kq+7
  f.q[1] = *(const uint4*)(base + 16);       // K = kq+16 .. kq+23
  return f.h;
}

// B stored TRANSPOSED in LDS: BT[n][k] (ldk = LDT). Lane holds col n = lane&15,
// K = kb..kb+7 and kb+16..kb+23 (matches A) => two b128 LDS loads.
DEVI v16h load_bT_frag(const _Float16* BT, int lane) {
  int n  = lane & 15;
  int kb = (lane >> 4) << 3;                 // 0 or 8
  const _Float16* p = BT + (size_t)n * LDT + kb;
  union { v16h h; uint4 q[2]; } f;
  f.q[0] = *(const uint4*)(p);
  f.q[1] = *(const uint4*)(p + 16);
  return f.h;
}

// Wave computes rows [16*wave,16*wave+16) x cols [0,128): 8 accumulator tiles.
// lA: row-major 128xLDT, lBT: transposed (col-major) 128xLDT.
DEVI void wmma_rowblock(const _Float16* lA, const _Float16* lBT,
                        int wave, int lane, v8f* acc) {
  for (int kc = 0; kc < 4; ++kc) {
    v16h a = load_a_frag(lA + (size_t)(wave << 4) * LDT + (kc << 5), LDT, lane);
#pragma unroll
    for (int t = 0; t < 8; ++t) {
      v16h b = load_bT_frag(lBT + (size_t)(t << 4) * LDT + (kc << 5), lane);
      acc[t] = __builtin_amdgcn_wmma_f32_16x16x32_f16(false, a, false, b,
                                                      (short)0, acc[t], false, false);
    }
  }
}

// Stage a 128x128-half tile global->LDS (row-padded to LDT), zero-filling invalid
// rows. srcStrideU4 = source row stride in uint4 units. Uses CDNA5 async LDS loads
// (ASYNCcnt) when available.
DEVI void stage_rows(_Float16* dst, const _Float16* src, int rowsValid,
                     int srcStrideU4, int tid) {
  uint4* d4 = (uint4*)dst;
  const uint4* s4 = (const uint4*)src;
  for (int i = tid; i < 2048; i += 256) {
    int r = i >> 4, c = i & 15;
#if HAVE_ASYNC_LDS
    if (r < rowsValid) {
      __builtin_amdgcn_global_load_async_to_lds_b128(
          (as1_i32x4)(s4 + (size_t)r * srcStrideU4 + c),
          (as3_i32x4)(d4 + (size_t)r * LDTU4 + c), 0, 0);
    } else {
      d4[(size_t)r * LDTU4 + c] = make_uint4(0u, 0u, 0u, 0u);
    }
#else
    uint4 v = make_uint4(0u, 0u, 0u, 0u);
    if (r < rowsValid) v = s4[(size_t)r * srcStrideU4 + c];
    d4[(size_t)r * LDTU4 + c] = v;
#endif
  }
}

DEVI void stage_fence() {
#if HAVE_ASYNC_LDS
  __builtin_amdgcn_s_wait_asynccnt(0);
#endif
}

// ---------------- small elementwise kernels ----------------
// Transposing f32 -> f16 weight conversion: d[n*K + k] = s[k*Nc + n]
__global__ void k_f32_to_f16_t(_Float16* __restrict__ d, const float* __restrict__ s,
                               int K, int Nc) {
  int i = blockIdx.x * 256 + threadIdx.x;
  if (i >= K * Nc) return;
  int n = i / K, k = i - n * K;
  d[i] = (_Float16)s[(size_t)k * Nc + n];
}

// LayerNorm of 128-channel rows, f32 -> f16. One warp per row.
__global__ __launch_bounds__(256) void k_ln_rows(const float* __restrict__ src,
                                                 const float* __restrict__ g,
                                                 const float* __restrict__ b,
                                                 _Float16* __restrict__ dst, int rows) {
  int wave = threadIdx.x >> 5, lane = threadIdx.x & 31;
  int row = blockIdx.x * 8 + wave;
  if (row >= rows) return;
  const float* p = src + (size_t)row * 128;
  float v0[4], s = 0.f, s2 = 0.f;
#pragma unroll
  for (int j = 0; j < 4; ++j) { float x = p[lane * 4 + j]; v0[j] = x; s += x; s2 += x * x; }
#pragma unroll
  for (int m = 1; m < 32; m <<= 1) { s += __shfl_xor(s, m, 32); s2 += __shfl_xor(s2, m, 32); }
  float mu  = s * (1.f / 128.f);
  float var = s2 * (1.f / 128.f) - mu * mu;
  float inv = rsqrtf(var + 1e-5f);
  _Float16* q = dst + (size_t)row * 128;
#pragma unroll
  for (int j = 0; j < 4; ++j) {
    int c = lane * 4 + j;
    q[c] = (_Float16)((v0[j] - mu) * inv * g[c] + b[c]);
  }
}

// Edge LN + pairwise distance.
__global__ __launch_bounds__(256) void k_ln_edge(const float* __restrict__ src,
                                                 const float* __restrict__ g,
                                                 const float* __restrict__ b,
                                                 const float* __restrict__ coords,
                                                 const int* __restrict__ eidx,
                                                 _Float16* __restrict__ dst,
                                                 float* __restrict__ distb, int E_) {
  int wave = threadIdx.x >> 5, lane = threadIdx.x & 31;
  int row = blockIdx.x * 8 + wave;
  if (row >= E_) return;
  const float* p = src + (size_t)row * 128;
  float v0[4], s = 0.f, s2 = 0.f;
#pragma unroll
  for (int j = 0; j < 4; ++j) { float x = p[lane * 4 + j]; v0[j] = x; s += x; s2 += x * x; }
#pragma unroll
  for (int m = 1; m < 32; m <<= 1) { s += __shfl_xor(s, m, 32); s2 += __shfl_xor(s2, m, 32); }
  float mu  = s * (1.f / 128.f);
  float var = s2 * (1.f / 128.f) - mu * mu;
  float inv = rsqrtf(var + 1e-5f);
  _Float16* q = dst + (size_t)row * 128;
#pragma unroll
  for (int j = 0; j < 4; ++j) {
    int c = lane * 4 + j;
    q[c] = (_Float16)((v0[j] - mu) * inv * g[c] + b[c]);
  }
  if (lane == 0) {
    int ir = eidx[row], ic = eidx[E_ + row];
    float dx = coords[ir * 3 + 0] - coords[ic * 3 + 0];
    float dy = coords[ir * 3 + 1] - coords[ic * 3 + 1];
    float dz = coords[ir * 3 + 2] - coords[ic * 3 + 2];
    distb[row] = 0.1f * sqrtf(dx * dx + dy * dy + dz * dz);
  }
}

// h_attn = wV / (z + 1e-6) -> f16
__global__ void k_hattn(const float* __restrict__ wv, const float* __restrict__ z,
                        _Float16* __restrict__ out, int n) {
  int i = blockIdx.x * 256 + threadIdx.x;
  if (i >= n) return;
  int c = i & 127, node = i >> 7, head = c >> 4;
  out[i] = (_Float16)(wv[i] / (z[(size_t)node * 8 + head] + 1e-6f));
}

// ---------------- q,k,v projections: one A-tile, three WMMA GEMMs ----------------
__global__ __launch_bounds__(256) void k_qkv(const _Float16* __restrict__ hln,
                                             const _Float16* __restrict__ wqT,
                                             const _Float16* __restrict__ wkT,
                                             const _Float16* __restrict__ wvT,
                                             float* __restrict__ Qf, float* __restrict__ Kf,
                                             float* __restrict__ Vf, int rows) {
  __shared__ __align__(16) _Float16 sA[128 * LDT];
  __shared__ __align__(16) _Float16 sB[128 * LDT];
  int tid = threadIdx.x, wave = tid >> 5, lane = tid & 31;
  int r0 = blockIdx.x * 128;
  int rowsValid = rows - r0; if (rowsValid > 128) rowsValid = 128;
  stage_rows(sA, hln + (size_t)r0 * 128, rowsValid, 16, tid);
  int hi = lane >> 4, nb = lane & 15;
  for (int m = 0; m < 3; ++m) {
    const _Float16* w = (m == 0) ? wqT : ((m == 1) ? wkT : wvT);
    float* o = (m == 0) ? Qf : ((m == 1) ? Kf : Vf);
    __syncthreads();
    stage_rows(sB, w, 128, 16, tid);
    stage_fence();
    __syncthreads();
    v8f acc[8] = {};
    wmma_rowblock(sA, sB, wave, lane, acc);
#pragma unroll
    for (int r = 0; r < 8; ++r) {
      int gr = r0 + (wave << 4) + r + (hi << 3);
      if (gr < rows) {
#pragma unroll
        for (int t = 0; t < 8; ++t) o[(size_t)gr * 128 + (t << 4) + nb] = acc[t][r];
      }
    }
  }
}

// ---------------- ep GEMM fused with attention + scatter ----------------
__global__ __launch_bounds__(256) void k_edge_attn(
    const _Float16* __restrict__ eln, const _Float16* __restrict__ weT,
    const float* __restrict__ welast, const float* __restrict__ dist,
    const int* __restrict__ eidx,
    const float* __restrict__ Qf, const float* __restrict__ Kf, const float* __restrict__ Vf,
    _Float16* __restrict__ alpha16, float* __restrict__ wv, float* __restrict__ zacc, int E_) {
  __shared__ __align__(16) _Float16 sA[128 * LDT];
  __shared__ __align__(16) _Float16 sB[128 * LDT];
  int tid = threadIdx.x, wave = tid >> 5, lane = tid & 31;
  int e0 = blockIdx.x * 128;
  int rowsValid = E_ - e0; if (rowsValid > 128) rowsValid = 128;
  stage_rows(sA, eln + (size_t)e0 * 128, rowsValid, 16, tid);
  stage_rows(sB, weT, 128, 16, tid);
  stage_fence();
  __syncthreads();
  v8f acc[8] = {};
  wmma_rowblock(sA, sB, wave, lane, acc);
  int hi = lane >> 4, nb = lane & 15;
#pragma unroll
  for (int r = 0; r < 8; ++r) {
    int ml = (wave << 4) + r + (hi << 3);
    int e  = e0 + ml;
    bool ok = (e < E_);
    int ec = ok ? e : 0;
    int ir = eidx[ec];
    int ic = eidx[E_ + ec];
    float dv = dist[ec];
    const float* qr = Qf + (size_t)ic * 128;
    const float* kr = Kf + (size_t)ir * 128;
    const float* vr = Vf + (size_t)ir * 128;
#pragma unroll
    for (int t = 0; t < 8; ++t) {               // t == head (Dh = 16 == tile width)
      int ch = (t << 4) + nb;
      float ep = acc[t][r] + dv * welast[ch];   // rank-1 fix for K=129
      float kq = kr[ch] * qr[ch] * 0.25f;       // 1/sqrt(16)
      kq = fminf(5.f, fmaxf(-5.f, kq));
      float al = kq * ep;
      if (ok) alpha16[(size_t)e * 128 + ch] = (_Float16)al;
      float s = al;                              // reduce over Dh within 16-lane half
      s += __shfl_xor(s, 1, 32);
      s += __shfl_xor(s, 2, 32);
      s += __shfl_xor(s, 4, 32);
      s += __shfl_xor(s, 8, 32);
      s = fminf(5.f, fmaxf(-5.f, s));
      float ax = expf(s);
      if (ok) {
        atomicAdd(&wv[(size_t)ic * 128 + ch], vr[ch] * ax);
        if (nb == 0) atomicAdd(&zacc[(size_t)ic * 8 + t], ax);
      }
    }
  }
}

// ---------------- output projection + bias + residual + LayerNorm2 ----------------
__global__ __launch_bounds__(256) void k_out_ln(
    const _Float16* __restrict__ a16, const _Float16* __restrict__ wT,
    const float* __restrict__ bias, const float* __restrict__ resid,
    const float* __restrict__ gam, const float* __restrict__ bet,
    float* __restrict__ out, _Float16* __restrict__ norm16, int rows) {
  __shared__ __align__(16) _Float16 sA[128 * LDT];
  __shared__ __align__(16) _Float16 sB[128 * LDT];
  int tid = threadIdx.x, wave = tid >> 5, lane = tid & 31;
  int r0 = blockIdx.x * 128;
  int rowsValid = rows - r0; if (rowsValid > 128) rowsValid = 128;
  stage_rows(sA, a16 + (size_t)r0 * 128, rowsValid, 16, tid);
  stage_rows(sB, wT, 128, 16, tid);
  stage_fence();
  __syncthreads();
  v8f acc[8] = {};
  wmma_rowblock(sA, sB, wave, lane, acc);
  int hi = lane >> 4, nb = lane & 15;
#pragma unroll
  for (int r = 0; r < 8; ++r) {
    int gr = r0 + (wave << 4) + r + (hi << 3);
    bool ok = (gr < rows);
    int gc = ok ? gr : 0;
    float vals[8], s = 0.f, s2 = 0.f;
#pragma unroll
    for (int t = 0; t < 8; ++t) {
      int ch = (t << 4) + nb;
      float x = acc[t][r] + bias[ch] + resid[(size_t)gc * 128 + ch];
      vals[t] = x; s += x; s2 += x * x;
    }
    s  += __shfl_xor(s, 1, 32);  s  += __shfl_xor(s, 2, 32);
    s  += __shfl_xor(s, 4, 32);  s  += __shfl_xor(s, 8, 32);
    s2 += __shfl_xor(s2, 1, 32); s2 += __shfl_xor(s2, 2, 32);
    s2 += __shfl_xor(s2, 4, 32); s2 += __shfl_xor(s2, 8, 32);
    float mu  = s * (1.f / 128.f);
    float var = s2 * (1.f / 128.f) - mu * mu;
    float inv = rsqrtf(var + 1e-5f);
    if (ok) {
#pragma unroll
      for (int t = 0; t < 8; ++t) {
        int ch = (t << 4) + nb;
        out[(size_t)gr * 128 + ch] = vals[t];
        norm16[(size_t)gr * 128 + ch] = (_Float16)((vals[t] - mu) * inv * gam[ch] + bet[ch]);
      }
    }
  }
}

// ---------------- FFN: out += silu(x@W1)@W2, hidden kept in LDS per 128-col half ----------------
__global__ __launch_bounds__(256) void k_ffn(const _Float16* __restrict__ norm16,
                                             const _Float16* __restrict__ w1T, // 256x128
                                             const _Float16* __restrict__ w2T, // 128x256
                                             float* __restrict__ out, int rows) {
  __shared__ __align__(16) _Float16 sA[128 * LDT];
  __shared__ __align__(16) _Float16 sB[128 * LDT];
  __shared__ __align__(16) _Float16 sH[128 * LDT];
  int tid = threadIdx.x, wave = tid >> 5, lane = tid & 31;
  int r0 = blockIdx.x * 128;
  int rowsValid = rows - r0; if (rowsValid > 128) rowsValid = 128;
  stage_rows(sA, norm16 + (size_t)r0 * 128, rowsValid, 16, tid);
  int hi = lane >> 4, nb = lane & 15;
  v8f acc2[8] = {};
  for (int half = 0; half < 2; ++half) {
    __syncthreads();
    // W1^T rows [half*128, half*128+128): contiguous 128x128 block, stride 16 uint4
    stage_rows(sB, w1T + (size_t)half * 128 * 128, 128, 16, tid);
    stage_fence();
    __syncthreads();
    v8f acc[8] = {};
    wmma_rowblock(sA, sB, wave, lane, acc);
#pragma unroll
    for (int r = 0; r < 8; ++r) {
      int ml = (wave << 4) + r + (hi << 3);
#pragma unroll
      for (int t = 0; t < 8; ++t) {
        float x = acc[t][r];
        float sl = x / (1.f + expf(-x));       // silu
        sH[(size_t)ml * LDT + (t << 4) + nb] = (_Float16)sl;
      }
    }
    __syncthreads();
    // W2^T k-half: rows n=0..127, k in [half*128, half*128+128), src stride 256 halves
    stage_rows(sB, w2T + (size_t)half * 128, 128, 32, tid);
    stage_fence();
    __syncthreads();
    wmma_rowblock(sH, sB, wave, lane, acc2);
  }
#pragma unroll
  for (int r = 0; r < 8; ++r) {
    int gr = r0 + (wave << 4) + r + (hi << 3);
    if (gr < rows) {
#pragma unroll
      for (int t = 0; t < 8; ++t) {
        size_t o = (size_t)gr * 128 + (t << 4) + nb;
        out[o] = out[o] + acc2[t][r];
      }
    }
  }
}

// ---------------- host orchestration ----------------
extern "C" void kernel_launch(void* const* d_in, const int* in_sizes, int n_in,
                              void* d_out, int out_size, void* d_ws, size_t ws_size,
                              hipStream_t stream) {
  (void)in_sizes; (void)n_in; (void)out_size; (void)ws_size;
  constexpr int N = 50000, E = 600000;
  constexpr int NBN = (N + 127) / 128;        // 391
  constexpr int NBE = (E + 127) / 128;        // 4688
  constexpr size_t NPAD = (size_t)NBN * 128;  // 50048
  constexpr size_t EPAD = (size_t)NBE * 128;  // 600064

  const float* node_feats = (const float*)d_in[0];
  const float* edge_feats = (const float*)d_in[1];
  const float* coords     = (const float*)d_in[2];
  const int*   eidx       = (const int*)d_in[3];
  const float* Wq   = (const float*)d_in[4];
  const float* Wk   = (const float*)d_in[5];
  const float* Wv   = (const float*)d_in[6];
  const float* We   = (const float*)d_in[7];
  const float* Wo_n = (const float*)d_in[8];
  const float* bo_n = (const float*)d_in[9];
  const float* Wo_e = (const float*)d_in[10];
  const float* bo_e = (const float*)d_in[11];
  const float* ln1n_g = (const float*)d_in[12];
  const float* ln1n_b = (const float*)d_in[13];
  const float* ln1e_g = (const float*)d_in[14];
  const float* ln1e_b = (const float*)d_in[15];
  const float* ln2n_g = (const float*)d_in[16];
  const float* ln2n_b = (const float*)d_in[17];
  const float* ln2e_g = (const float*)d_in[18];
  const float* ln2e_b = (const float*)d_in[19];
  const float* Wn1 = (const float*)d_in[20];
  const float* Wn2 = (const float*)d_in[21];
  const float* We1 = (const float*)d_in[22];
  const float* We2 = (const float*)d_in[23];

  char* ws = (char*)d_ws;
  size_t off = 0;
  auto take = [&](size_t bytes) { size_t r = off; off += (bytes + 255) & ~(size_t)255; return r; };

  // all weights stored TRANSPOSED f16: WT[n*K + k]
  _Float16* WQ16  = (_Float16*)(ws + take(16384 * 2));
  _Float16* WK16  = (_Float16*)(ws + take(16384 * 2));
  _Float16* WV16  = (_Float16*)(ws + take(16384 * 2));
  _Float16* WE16  = (_Float16*)(ws + take(16384 * 2));
  _Float16* WON16 = (_Float16*)(ws + take(16384 * 2));
  _Float16* WOE16 = (_Float16*)(ws + take(16384 * 2));
  _Float16* WN1_16 = (_Float16*)(ws + take(32768 * 2));  // 256x128
  _Float16* WN2_16 = (_Float16*)(ws + take(32768 * 2));  // 128x256
  _Float16* WE1_16 = (_Float16*)(ws + take(32768 * 2));
  _Float16* WE2_16 = (_Float16*)(ws + take(32768 * 2));
  float*    WELAST = (float*)(ws + take(128 * 4));
  _Float16* HLN16  = (_Float16*)(ws + take(NPAD * 128 * 2));  // later reused as h_attn f16
  _Float16* ELN16  = (_Float16*)(ws + take(EPAD * 128 * 2));  // later reused as en f16
  float*    DIST   = (float*)(ws + take(EPAD * 4));
  float*    Qf     = (float*)(ws + take(NPAD * 128 * 4));     // later reused as hn f16
  float*    Kf     = (float*)(ws + take(NPAD * 128 * 4));
  float*    Vf     = (float*)(ws + take(NPAD * 128 * 4));
  _Float16* ALPHA16 = (_Float16*)(ws + take(EPAD * 128 * 2));
  float*    WVf    = (float*)(ws + take((NPAD * 128 + NPAD * 8) * 4));
  float*    Zf     = WVf + NPAD * 128;
  _Float16* HATT16 = HLN16;              // reuse: HLN consumed by k_qkv
  _Float16* EN16   = ELN16;              // reuse: ELN consumed by k_edge_attn
  _Float16* HN16   = (_Float16*)Qf;      // reuse: Q consumed by k_edge_attn

  float* outN = (float*)d_out;
  float* outE = outN + (size_t)N * 128;

  // zero the scatter accumulators (capture-safe)
  (void)hipMemsetAsync(WVf, 0, (NPAD * 128 + NPAD * 8) * sizeof(float), stream);

  // weight conversion f32 -> f16 with transpose
  auto cvtT = [&](const float* s, _Float16* d, int K, int Nc) {
    int n = K * Nc;
    k_f32_to_f16_t<<<(n + 255) / 256, 256, 0, stream>>>(d, s, K, Nc);
  };
  cvtT(Wq, WQ16, 128, 128);   cvtT(Wk, WK16, 128, 128);   cvtT(Wv, WV16, 128, 128);
  cvtT(We, WE16, 128, 128);   // first 128 rows of the 129x128 We
  cvtT(Wo_n, WON16, 128, 128); cvtT(Wo_e, WOE16, 128, 128);
  cvtT(Wn1, WN1_16, 128, 256); cvtT(Wn2, WN2_16, 256, 128);
  cvtT(We1, WE1_16, 128, 256); cvtT(We2, WE2_16, 256, 128);
  (void)hipMemcpyAsync(WELAST, We + 16384, 128 * sizeof(float),
                       hipMemcpyDeviceToDevice, stream);

  // LN1
  k_ln_rows<<<(N + 7) / 8, 256, 0, stream>>>(node_feats, ln1n_g, ln1n_b, HLN16, N);
  k_ln_edge<<<(E + 7) / 8, 256, 0, stream>>>(edge_feats, ln1e_g, ln1e_b, coords, eidx,
                                             ELN16, DIST, E);
  // q,k,v
  k_qkv<<<NBN, 256, 0, stream>>>(HLN16, WQ16, WK16, WV16, Qf, Kf, Vf, N);
  // ep GEMM + alpha + alphax + scatter
  k_edge_attn<<<NBE, 256, 0, stream>>>(ELN16, WE16, WELAST, DIST, eidx, Qf, Kf, Vf,
                                       ALPHA16, WVf, Zf, E);
  // h_attn = wV/(z+1e-6)
  k_hattn<<<(N * 128 + 255) / 256, 256, 0, stream>>>(WVf, Zf, HATT16, N * 128);
  // output projections + residual + LN2  (fully overwrite d_out)
  k_out_ln<<<NBN, 256, 0, stream>>>(HATT16, WON16, bo_n, node_feats, ln2n_g, ln2n_b,
                                    outN, HN16, N);
  k_out_ln<<<NBE, 256, 0, stream>>>(ALPHA16, WOE16, bo_e, edge_feats, ln2e_g, ln2e_b,
                                    outE, EN16, E);
  // FFNs (read h2/e2 from d_out, write h2+h3 / e2+e3)
  k_ffn<<<NBN, 256, 0, stream>>>(HN16, WN1_16, WN2_16, outN, N);
  k_ffn<<<NBE, 256, 0, stream>>>(EN16, WE1_16, WE2_16, outE, E);
}
